// TkLastTradesAutoencoder_19344532702124
// MI455X (gfx1250) — matrix operand order repaired and checked by reading, not compiled, not run on hardware
//
#include <hip/hip_runtime.h>
#include <hip/hip_bf16.h>

// ---------------------------------------------------------------------------
// CDNA5 (gfx1250) k-sparse autoencoder forward pass.
//   - fp32 weights/x pre-converted to bf16 (HBM-bound, ~15us)
//   - 4 GEMMs on v_wmma_f32_16x16x32_bf16, double-buffered LDS,
//     async global->LDS copy (GLOBAL_LOAD_ASYNC_TO_LDS_B128) for A tiles,
//     padded LDS (conflict-free b128 fragment reads), software-pipelined
//     B-fragment loads, fused bias/relu/sigmoid epilogues
//   - per-row top-64 threshold select, 32-wide softmax
// ---------------------------------------------------------------------------

#define USE_ASYNC_A 1   // gfx1250 GLOBAL_LOAD_ASYNC_TO_LDS_B128 path

typedef __attribute__((ext_vector_type(16))) __bf16 v16bf;
typedef __attribute__((ext_vector_type(8)))  float  v8f;

__device__ __forceinline__ unsigned short f2bf(float f) {
  union { float f; unsigned u; } c; c.f = f;
  unsigned u = c.u;
  u += 0x7FFFu + ((u >> 16) & 1u);   // round-to-nearest-even
  return (unsigned short)(u >> 16);
}

union FragAB { unsigned u[8]; v16bf v; };

__device__ __forceinline__ void async_copy_b128(const void* gsrc, void* ldst) {
#if defined(__AMDGCN__)
  unsigned loff = (unsigned)(unsigned long long)(uintptr_t)ldst;  // low 32b = LDS offset
  asm volatile("global_load_async_to_lds_b128 %0, %1, off"
               :: "v"(loff), "v"(gsrc) : "memory");
#else
  (void)gsrc; (void)ldst;
#endif
}
__device__ __forceinline__ void wait_async0() {
#if defined(__AMDGCN__)
  asm volatile("s_wait_asynccnt 0x0" ::: "memory");
#endif
}

// ---------------------------------------------------------------------------
// fp32 -> bf16 bulk conversion (grid-stride, 8 elems/thread, b128 stores)
// ---------------------------------------------------------------------------
__global__ __launch_bounds__(256)
void f32_to_bf16_kernel(const float* __restrict__ src,
                        unsigned short* __restrict__ dst, long long n)
{
  long long i = ((long long)blockIdx.x * 256 + threadIdx.x) * 8;
  const long long stride = (long long)gridDim.x * 256 * 8;
  for (; i < n; i += stride) {
    float4 f0 = *(const float4*)(src + i);
    float4 f1 = *(const float4*)(src + i + 4);
    unsigned short r[8];
    r[0] = f2bf(f0.x); r[1] = f2bf(f0.y); r[2] = f2bf(f0.z); r[3] = f2bf(f0.w);
    r[4] = f2bf(f1.x); r[5] = f2bf(f1.y); r[6] = f2bf(f1.z); r[7] = f2bf(f1.w);
    *(uint4*)(dst + i) = *(const uint4*)r;
  }
}

// ---------------------------------------------------------------------------
// Tiled GEMM: C[M,N] = act(A[M,K] @ B[K,N] + bias[N]),  A,B bf16 row-major.
// Block 256 threads (8 waves), tile 128x128, K-step 32, double-buffered LDS.
// LDS row stride 40 ushorts (80B): conflict-free b128 fragment reads.
// ---------------------------------------------------------------------------
template<int ACT, bool OUT_BF16>
__global__ __launch_bounds__(256)
void gemm_bf16_wmma(const unsigned short* __restrict__ Ap,
                    const unsigned short* __restrict__ Bp,
                    const float* __restrict__ bias,
                    void* __restrict__ Cp,
                    int M, int N, int K)
{
  constexpr int LSTR = 40;
  __shared__ alignas(16) unsigned short lA[2][128 * LSTR]; // [m][k] bf16
  __shared__ alignas(16) unsigned short lB[2][128 * LSTR]; // [n][k] bf16 (transposed)

  const int tid   = threadIdx.x;
  const int lane  = tid & 31;
  const int wave  = tid >> 5;
  const int lhalf = (lane >> 4) & 1;
  const int l16   = lane & 15;

  const int n0 = blockIdx.x * 128;
  const int m0 = blockIdx.y * 128;

  v8f acc[8];
#pragma unroll
  for (int t = 0; t < 8; ++t)
#pragma unroll
    for (int r = 0; r < 8; ++r) acc[t][r] = 0.0f;

  // staging coordinates
  const int bkp = tid >> 4;          // 0..15  -> k rows {2bkp, 2bkp+1}
  const int bn8 = (tid & 15) * 8;    // 8 consecutive n

  const int Kt = K >> 5;

  auto stageA = [&](int kt, int buf) {
    const int k0 = kt << 5;
#pragma unroll
    for (int h = 0; h < 2; ++h) {
      const int c = tid + h * 256;          // 512 chunks of 8 ushorts
      const int row = c >> 2, ko = (c & 3) * 8;
      const unsigned short* s = Ap + (size_t)(m0 + row) * K + k0 + ko;
      unsigned short* d = &lA[buf][row * LSTR + ko];
#if USE_ASYNC_A
      async_copy_b128(s, d);
#else
      *(uint4*)d = *(const uint4*)s;
#endif
    }
  };

  auto loadB = [&](int kt, uint4& r0, uint4& r1) {
    const unsigned short* s = Bp + (size_t)((kt << 5) + 2 * bkp) * N + n0 + bn8;
    r0 = *(const uint4*)s;
    r1 = *(const uint4*)(s + N);
  };

  auto storeB = [&](int buf, const uint4& r0, const uint4& r1) {
    unsigned short* d = lB[buf];
    const unsigned* p0 = (const unsigned*)&r0;
    const unsigned* p1 = (const unsigned*)&r1;
#pragma unroll
    for (int i = 0; i < 4; ++i) {
      const unsigned lo = p0[i], hi = p1[i];
      const unsigned e = (lo & 0xFFFFu) | (hi << 16);        // n = bn8+2i
      const unsigned o = (lo >> 16) | (hi & 0xFFFF0000u);    // n = bn8+2i+1
      *(unsigned*)&d[(bn8 + 2 * i    ) * LSTR + 2 * bkp] = e;
      *(unsigned*)&d[(bn8 + 2 * i + 1) * LSTR + 2 * bkp] = o;
    }
  };

  const int mrow = wave * 16 + l16;
  const int aoff = lhalf * 8;    // A: lanes 16-31 start at k=8 / k=24
  const int boff = lhalf * 8;    // B: lanes 16-31 start at k=8 / k=24

  auto loadBfrag = [&](int buf, int t, FragAB& b) {
    const unsigned short* base = &lB[buf][(t * 16 + l16) * LSTR + boff];
    uint4 q0 = *(const uint4*)(base);
    uint4 q1 = *(const uint4*)(base + 16);
    b.u[0] = q0.x; b.u[1] = q0.y; b.u[2] = q0.z; b.u[3] = q0.w;
    b.u[4] = q1.x; b.u[5] = q1.y; b.u[6] = q1.z; b.u[7] = q1.w;
  };

  auto compute = [&](int buf) {
    FragAB a;
    {
      const unsigned short* base = &lA[buf][mrow * LSTR + aoff];
      uint4 q0 = *(const uint4*)(base);        // k pairs 0..7   (+aoff)
      uint4 q1 = *(const uint4*)(base + 16);   // k pairs 16..23 (+aoff)
      a.u[0] = q0.x; a.u[1] = q0.y; a.u[2] = q0.z; a.u[3] = q0.w;
      a.u[4] = q1.x; a.u[5] = q1.y; a.u[6] = q1.z; a.u[7] = q1.w;
    }
    // software-pipelined B fragments: load t+1 while WMMA t executes
    FragAB bping, bpong;
    loadBfrag(buf, 0, bping);
#pragma unroll
    for (int t = 0; t < 8; ++t) {
      FragAB& cur = (t & 1) ? bpong : bping;
      FragAB& nxt = (t & 1) ? bping : bpong;
      if (t < 7) loadBfrag(buf, t + 1, nxt);
      acc[t] = __builtin_amdgcn_wmma_f32_16x16x32_bf16(
          false, a.v, false, cur.v, (short)0, acc[t], false, false);
    }
  };

  // prologue: stage tile 0 into buffer 0
  {
    uint4 b0, b1;
    loadB(0, b0, b1);
    stageA(0, 0);
    storeB(0, b0, b1);
#if USE_ASYNC_A
    wait_async0();
#endif
    __syncthreads();
  }

  for (int kt = 0; kt < Kt; ++kt) {
    const int cur = kt & 1;
    const bool more = (kt + 1 < Kt);
    uint4 b0, b1;
    if (more) {                 // issue next tile's global traffic first
      stageA(kt + 1, cur ^ 1);  // async DMA into other buffer
      loadB(kt + 1, b0, b1);    // global loads in flight during compute
    }
    compute(cur);
    if (more) {
      storeB(cur ^ 1, b0, b1);
#if USE_ASYNC_A
      wait_async0();
#endif
    }
    __syncthreads();
  }

  // epilogue: bias + activation + store (C/D: VGPR r -> row r+8*lhalf, col l16)
#pragma unroll
  for (int t = 0; t < 8; ++t) {
    const int n = n0 + t * 16 + l16;
    const float bv = bias ? bias[n] : 0.0f;
#pragma unroll
    for (int r = 0; r < 8; ++r) {
      const int m = m0 + wave * 16 + lhalf * 8 + r;
      float v = acc[t][r] + bv;
      if (ACT == 1) v = fmaxf(v, 0.0f);
      if (ACT == 2) v = 1.0f / (1.0f + __expf(-v));
      if (OUT_BF16) ((unsigned short*)Cp)[(size_t)m * N + n] = f2bf(v);
      else          ((float*)Cp)[(size_t)m * N + n] = v;
    }
  }
}

// ---------------------------------------------------------------------------
// Per-row top-k (k=64): binary-search threshold in LDS, deterministic ties.
// ---------------------------------------------------------------------------
__global__ __launch_bounds__(256)
void topk_mask_kernel(const float* __restrict__ code,
                      unsigned short* __restrict__ out,
                      int D, int k)
{
  __shared__ float vals[2048];
  __shared__ unsigned char flags[2048];
  __shared__ int cnt;

  const int row = blockIdx.x;
  const float* rp = code + (size_t)row * D;

  for (int i = threadIdx.x; i < D; i += blockDim.x) { vals[i] = rp[i]; flags[i] = 0; }
  __syncthreads();

  float lo = 0.0f, hi = 1.0f;   // invariant: count(>lo) >= k, count(>hi) < k
  for (int it = 0; it < 30; ++it) {
    const float mid = 0.5f * (lo + hi);
    if (threadIdx.x == 0) cnt = 0;
    __syncthreads();
    int c = 0;
    for (int i = threadIdx.x; i < D; i += blockDim.x) c += (vals[i] > mid);
    atomicAdd(&cnt, c);
    __syncthreads();
    const int total = cnt;
    __syncthreads();
    if (total >= k) lo = mid; else hi = mid;
  }

  if (threadIdx.x == 0) cnt = 0;
  __syncthreads();
  int c = 0;
  for (int i = threadIdx.x; i < D; i += blockDim.x)
    if (vals[i] > hi) { flags[i] = 1; ++c; }
  atomicAdd(&cnt, c);
  __syncthreads();
  if (threadIdx.x == 0) {
    int need = k - cnt;
    for (int i = 0; i < D && need > 0; ++i)
      if (!flags[i] && vals[i] > lo) { flags[i] = 1; --need; }
  }
  __syncthreads();

  for (int i = threadIdx.x; i < D; i += blockDim.x)
    out[(size_t)row * D + i] = flags[i] ? f2bf(vals[i]) : (unsigned short)0;
}

// ---------------------------------------------------------------------------
// In-place softmax over contiguous groups of 32 (vocab dim).
// ---------------------------------------------------------------------------
__global__ __launch_bounds__(256)
void softmax32_kernel(float* __restrict__ out, int ngroups)
{
  const int g = blockIdx.x * blockDim.x + threadIdx.x;
  if (g >= ngroups) return;
  float* p = out + (size_t)g * 32;
  float m = -3.4e38f;
#pragma unroll
  for (int i = 0; i < 32; ++i) m = fmaxf(m, p[i]);
  float e[32]; float s = 0.0f;
#pragma unroll
  for (int i = 0; i < 32; ++i) { e[i] = __expf(p[i] - m); s += e[i]; }
  const float inv = 1.0f / s;
#pragma unroll
  for (int i = 0; i < 32; ++i) p[i] = e[i] * inv;
}

// ---------------------------------------------------------------------------
extern "C" void kernel_launch(void* const* d_in, const int* in_sizes, int n_in,
                              void* d_out, int out_size, void* d_ws, size_t ws_size,
                              hipStream_t stream)
{
  const float* x   = (const float*)d_in[0];
  const float* We1 = (const float*)d_in[1];
  const float* be1 = (const float*)d_in[2];
  const float* We2 = (const float*)d_in[3];
  const float* be2 = (const float*)d_in[4];
  const float* Wd1 = (const float*)d_in[5];
  const float* bd1 = (const float*)d_in[6];
  const float* Wd2 = (const float*)d_in[7];
  const float* bd2 = (const float*)d_in[8];

  const int D_H    = in_sizes[2];                 // 4096
  const int D_CODE = in_sizes[4];                 // 2048
  const int D_OUT  = in_sizes[8];                 // 4096 (= 128*32)
  const int D_IN   = in_sizes[1] / D_H;           // 4096
  const int B      = in_sizes[0] / D_IN;          // 2048
  const int K_SPARSE = 64;
  (void)n_in; (void)out_size; (void)ws_size;

  // workspace regions (reused once dead; all launches stream-ordered):
  //   R0: We1b -> Wd2b          (D_IN*D_H*2 bytes, largest weight)
  //   R1: xb -> We2b -> Wd1b    (B*D_IN*2 == D_H*D_CODE*2 == D_CODE*D_H*2)
  //   R2: h -> h2               (B*D_H*2)
  //   R3: code f32              (B*D_CODE*4)
  //   R4: codem bf16            (B*D_CODE*2)
  char* ws = (char*)d_ws;
  const size_t szR0 = (size_t)D_IN * D_H * 2;
  const size_t szR1 = (size_t)B * D_IN * 2;
  const size_t szR2 = (size_t)B * D_H * 2;
  const size_t szR3 = (size_t)B * D_CODE * 4;
  unsigned short* R0 = (unsigned short*)(ws);
  unsigned short* R1 = (unsigned short*)(ws + szR0);
  unsigned short* R2 = (unsigned short*)(ws + szR0 + szR1);
  float*          R3 = (float*)(ws + szR0 + szR1 + szR2);
  unsigned short* R4 = (unsigned short*)(ws + szR0 + szR1 + szR2 + szR3);

  const dim3 blk(256);
  auto cvt = [&](const float* s, unsigned short* d, long long n) {
    long long want = (n / 8 + 255) / 256;
    int blocks = (int)(want < 4096 ? want : 4096);
    f32_to_bf16_kernel<<<dim3(blocks), blk, 0, stream>>>(s, d, n);
  };

  // 0) bf16 conversions interleaved with consumers via stream order
  cvt(x,   R1, (long long)B * D_IN);       // xb
  cvt(We1, R0, (long long)D_IN * D_H);     // We1b

  // 1) h = relu(xb @ We1b + be1)
  gemm_bf16_wmma<1, true><<<dim3(D_H / 128, B / 128), blk, 0, stream>>>(
      R1, R0, be1, R2, B, D_H, D_IN);

  cvt(We2, R1, (long long)D_H * D_CODE);   // We2b (xb dead)

  // 2) code = sigmoid(h @ We2b + be2)
  gemm_bf16_wmma<2, false><<<dim3(D_CODE / 128, B / 128), blk, 0, stream>>>(
      R2, R1, be2, R3, B, D_CODE, D_H);

  // 3) top-64 mask -> codem bf16
  topk_mask_kernel<<<dim3(B), blk, 0, stream>>>(R3, R4, D_CODE, K_SPARSE);

  cvt(Wd1, R1, (long long)D_CODE * D_H);   // Wd1b (We2b dead)

  // 4) h2 = relu(codem @ Wd1b + bd1)   (h dead -> R2 reused)
  gemm_bf16_wmma<1, true><<<dim3(D_H / 128, B / 128), blk, 0, stream>>>(
      R4, R1, bd1, R2, B, D_H, D_CODE);

  cvt(Wd2, R0, (long long)D_H * D_OUT);    // Wd2b (We1b dead)

  // 5) logits = h2 @ Wd2b + bd2 -> d_out
  gemm_bf16_wmma<0, false><<<dim3(D_OUT / 128, B / 128), blk, 0, stream>>>(
      R2, R0, bd2, (float*)d_out, B, D_OUT, D_H);

  // 6) softmax over vocab dim (groups of 32), in place
  const int ngroups = (B * D_OUT) / 32;
  softmax32_kernel<<<dim3((ngroups + 255) / 256), blk, 0, stream>>>(
      (float*)d_out, ngroups);
}
